// Model_39676907885324
// MI455X (gfx1250) — matrix operand order repaired and checked by reading, not compiled, not run on hardware
//
#include <hip/hip_runtime.h>
#include <hip/hip_bf16.h>

typedef __attribute__((ext_vector_type(16))) __bf16 bf16x16;
typedef __attribute__((ext_vector_type(8)))  __bf16 bf16x8;
typedef __attribute__((ext_vector_type(8)))  float  f32x8;

#define DD 1024
#define SS 2048
#define BB 4

// Lowers to v_cvt_pk_bf16_f32 (verified in round-2/3 disasm).
static __device__ inline __bf16 f2bf(float f) { return (__bf16)f; }

static __device__ inline f32x8 wmma_bf16(bf16x16 a, bf16x16 b, f32x8 c) {
    // D = A(16x32 bf16) * B(32x16 bf16) + C(16x16 f32)
    return __builtin_amdgcn_wmma_f32_16x16x32_bf16(false, a, false, b,
                                                   (short)0, c, false, false);
}

// -------------------------------------------------------------------------
// Kernel 0: one-shot fp32 -> bf16 conversion for the weight matrices
// (otherwise re-converted 64x by the GEMM M-blocks).
// -------------------------------------------------------------------------
__global__ void cvt_f32_bf16(const float* __restrict__ src, __bf16* __restrict__ dst,
                             int n) {
    const int i = (blockIdx.x * blockDim.x + threadIdx.x) * 8;
    if (i + 8 <= n) {
        f32x8 x = *reinterpret_cast<const f32x8*>(src + i);
        bf16x8 o;
#pragma unroll
        for (int h = 0; h < 8; ++h) o[h] = f2bf(x[h]);
        *reinterpret_cast<bf16x8*>(dst + i) = o;
    }
}

// -------------------------------------------------------------------------
// Kernel 1: projections  P = X @ W^T + bias   (X fp32, W pre-converted bf16)
//   z = 0: Qp row-major [B*S][D]
//   z = 1: Kp row-major [B*S][D]
//   z = 2: VpT transposed [B][D][S]
// Each wave: 16 rows x 64 cols, K-loop of 32 chunks. All four B fragments
// are loaded before the WMMA sequence so the loads form one clause and the
// WMMAs drain on partial loadcnt waits.
// -------------------------------------------------------------------------
__global__ void proj_kernel(const float* __restrict__ q, const float* __restrict__ k,
                            const float* __restrict__ v,
                            const __bf16* __restrict__ WqB, const float* __restrict__ bq,
                            const __bf16* __restrict__ WkB, const float* __restrict__ bk,
                            const __bf16* __restrict__ WvB, const float* __restrict__ bv,
                            __bf16* __restrict__ Qp, __bf16* __restrict__ Kp,
                            __bf16* __restrict__ VpT) {
    const int z = blockIdx.z;
    const float*  X    = (z == 0) ? q   : (z == 1) ? k   : v;
    const __bf16* W    = (z == 0) ? WqB : (z == 1) ? WkB : WvB;
    const float*  bias = (z == 0) ? bq  : (z == 1) ? bk  : bv;

    const int tid   = threadIdx.x;
    const int wave  = tid >> 5;
    const int lane  = tid & 31;
    const int lhalf = lane >> 4;   // 0 or 1
    const int l16   = lane & 15;

    const int m0 = blockIdx.x * 128 + wave * 16;   // row tile base
    const int n0 = blockIdx.y * 64;                // col tile base

    f32x8 acc[4];
#pragma unroll
    for (int nt = 0; nt < 4; ++nt)
#pragma unroll
        for (int r = 0; r < 8; ++r) acc[nt][r] = 0.0f;

    const float* xrow = X + (size_t)(m0 + l16) * DD + lhalf * 16;

#pragma unroll 2
    for (int kc = 0; kc < 32; ++kc) {
        // A fragment: rows = lane%16, contiguous K (k = 16*lhalf + s), same order as B
        f32x8 xlo = *reinterpret_cast<const f32x8*>(xrow + kc * 32);
        f32x8 xhi = *reinterpret_cast<const f32x8*>(xrow + kc * 32 + 8);
        bf16x16 a;
#pragma unroll
        for (int h = 0; h < 8; ++h) { a[h] = f2bf(xlo[h]); a[8 + h] = f2bf(xhi[h]); }

        // issue all 4 B-fragment loads first (one clause), then drain WMMAs
        bf16x16 bfr[4];
#pragma unroll
        for (int nt = 0; nt < 4; ++nt) {
            const __bf16* wrow = W + (size_t)(n0 + nt * 16 + l16) * DD
                                 + kc * 32 + lhalf * 16;
            bfr[nt] = *reinterpret_cast<const bf16x16*>(wrow);
        }
#pragma unroll
        for (int nt = 0; nt < 4; ++nt)
            acc[nt] = wmma_bf16(a, bfr[nt], acc[nt]);
    }

#pragma unroll
    for (int nt = 0; nt < 4; ++nt) {
        const int n = n0 + nt * 16 + l16;
        const float bb = bias[n];
#pragma unroll
        for (int r = 0; r < 8; ++r) {
            const float val = acc[nt][r] + bb;
            const int m = m0 + r + lhalf * 8;
            if (z == 0) {
                Qp[(size_t)m * DD + n] = f2bf(val);
            } else if (z == 1) {
                Kp[(size_t)m * DD + n] = f2bf(val);
            } else {
                const int bidx = m >> 11;          // / 2048
                const int s    = m & (SS - 1);
                VpT[((size_t)(bidx * DD + n)) * SS + s] = f2bf(val);
            }
        }
    }
}

// -------------------------------------------------------------------------
// Kernel 2: flash attention over projected bf16 tensors.
// Block = 16 queries, 256 threads (8 wave32). Per 128-key block:
//   each wave computes a 16x16 score panel (32 WMMAs over D, fully unrolled),
//   workgroup softmax (online max/sum) via LDS,
//   each wave accumulates 128 output columns of P@V (32 WMMAs).
// -------------------------------------------------------------------------
__global__ void attn_kernel(const __bf16* __restrict__ Qp, const __bf16* __restrict__ Kp,
                            const __bf16* __restrict__ VpT, float* __restrict__ Out) {
    __shared__ __bf16 q_lds[16 * DD];      // 32 KB
    __shared__ float  s_lds[16 * 128];     // 8 KB
    __shared__ __bf16 p_lds[16 * 128];     // 4 KB
    __shared__ float  red[16 * 8];
    __shared__ float  m_s[16], l_s[16], scale_s[16];

    const int tid   = threadIdx.x;
    const int wave  = tid >> 5;
    const int lane  = tid & 31;
    const int lhalf = lane >> 4;
    const int l16   = lane & 15;
    const int b     = blockIdx.y;
    const int qbase = blockIdx.x * 16;

    // ---- stage Q tile into LDS with gfx1250 async copies (ASYNCcnt path).
    // Flat pointers to LDS carry the LDS byte offset in their low 32 bits
    // (ISA 10.2 aperture mapping), so we can form per-lane LDS addresses.
    {
        const char* src = reinterpret_cast<const char*>(Qp + (size_t)(b * SS + qbase) * DD);
        const unsigned int lds_base = (unsigned int)(uintptr_t)(void*)q_lds;
        for (int i = tid; i < (16 * DD * 2) / 16; i += 256) {
            unsigned long long ga = (unsigned long long)(uintptr_t)(src + i * 16);
            unsigned int       la = lds_base + i * 16;
            asm volatile("global_load_async_to_lds_b128 %0, %1, off"
                         :: "v"(la), "v"(ga) : "memory");
        }
        asm volatile("s_wait_asynccnt 0x0" ::: "memory");
    }
    if (tid < 16) { m_s[tid] = -INFINITY; l_s[tid] = 0.0f; }
    __syncthreads();

    f32x8 acc[8];
#pragma unroll
    for (int ct = 0; ct < 8; ++ct)
#pragma unroll
        for (int r = 0; r < 8; ++r) acc[ct][r] = 0.0f;

    const float inv_scale = 0.03125f;   // 1/sqrt(1024)

    for (int kb = 0; kb < SS / 128; ++kb) {
        const int k0 = kb * 128;
        const __bf16* krow = Kp + (size_t)(b * SS + k0 + wave * 16 + l16) * DD + lhalf * 16;
        if (kb + 1 < SS / 128)
            __builtin_prefetch(krow + (size_t)128 * DD, 0, 1);   // next k-block K rows

        // ---- scores: 16x16 panel per wave, K-loop over D in 32-wide chunks
        f32x8 sc;
#pragma unroll
        for (int r = 0; r < 8; ++r) sc[r] = 0.0f;
#pragma unroll
        for (int dc = 0; dc < 32; ++dc) {
            const __bf16* qrow = q_lds + l16 * DD + dc * 32 + lhalf * 16;
            bf16x8 alo = *reinterpret_cast<const bf16x8*>(qrow);
            bf16x8 ahi = *reinterpret_cast<const bf16x8*>(qrow + 8);
            bf16x16 a;
#pragma unroll
            for (int h = 0; h < 8; ++h) { a[h] = alo[h]; a[8 + h] = ahi[h]; }
            bf16x16 bfrag = *reinterpret_cast<const bf16x16*>(krow + dc * 32);
            sc = wmma_bf16(a, bfrag, sc);
        }
#pragma unroll
        for (int r = 0; r < 8; ++r)
            s_lds[(r + lhalf * 8) * 128 + wave * 16 + l16] = sc[r] * inv_scale;
        __syncthreads();

        // ---- online softmax over the 16x128 panel
        if (tid < 128) {
            const int row = tid >> 3, seg = tid & 7;
            const float* sp = s_lds + row * 128 + seg * 16;
            float lm = -INFINITY;
#pragma unroll
            for (int j = 0; j < 16; ++j) lm = fmaxf(lm, sp[j]);
            red[row * 8 + seg] = lm;
        }
        __syncthreads();
        if (tid < 16) {
            const float m_old = m_s[tid];
            float mnew = m_old;
#pragma unroll
            for (int j = 0; j < 8; ++j) mnew = fmaxf(mnew, red[tid * 8 + j]);
            m_s[tid] = mnew;
            scale_s[tid] = __expf(m_old - mnew);
        }
        __syncthreads();
        if (tid < 128) {
            const int row = tid >> 3, seg = tid & 7;
            const float mnew = m_s[row];
            const float* sp = s_lds + row * 128 + seg * 16;
            float ls = 0.0f;
#pragma unroll
            for (int j = 0; j < 16; ++j) {
                const float p = __expf(sp[j] - mnew);
                p_lds[row * 128 + seg * 16 + j] = f2bf(p);
                ls += p;
            }
            red[row * 8 + seg] = ls;
        }
        __syncthreads();
        if (tid < 16) {
            float ls = 0.0f;
#pragma unroll
            for (int j = 0; j < 8; ++j) ls += red[tid * 8 + j];
            l_s[tid] = l_s[tid] * scale_s[tid] + ls;
        }
        __syncthreads();

        // ---- rescale accumulators, then P @ V
        float fsc[8];
#pragma unroll
        for (int r = 0; r < 8; ++r) fsc[r] = scale_s[r + lhalf * 8];
#pragma unroll
        for (int ct = 0; ct < 8; ++ct)
#pragma unroll
            for (int r = 0; r < 8; ++r) acc[ct][r] *= fsc[r];

#pragma unroll
        for (int kc = 0; kc < 4; ++kc) {
            const __bf16* prow = p_lds + l16 * 128 + kc * 32 + lhalf * 16;
            bf16x8 plo = *reinterpret_cast<const bf16x8*>(prow);
            bf16x8 phi = *reinterpret_cast<const bf16x8*>(prow + 8);
            bf16x16 a;
#pragma unroll
            for (int h = 0; h < 8; ++h) { a[h] = plo[h]; a[8 + h] = phi[h]; }
#pragma unroll
            for (int ct = 0; ct < 8; ++ct) {
                const int d = wave * 128 + ct * 16 + l16;
                const __bf16* vptr = VpT + ((size_t)(b * DD + d)) * SS
                                     + k0 + kc * 32 + lhalf * 16;
                bf16x16 bfrag = *reinterpret_cast<const bf16x16*>(vptr);
                acc[ct] = wmma_bf16(a, bfrag, acc[ct]);
            }
        }
        // next iteration's s_lds write is fenced by the __syncthreads() above
    }

    // ---- epilogue: divide by l, write fp32
    float linv[8];
#pragma unroll
    for (int r = 0; r < 8; ++r) linv[r] = 1.0f / l_s[r + lhalf * 8];
#pragma unroll
    for (int ct = 0; ct < 8; ++ct) {
#pragma unroll
        for (int r = 0; r < 8; ++r) {
            const int qrow = qbase + r + lhalf * 8;
            const int col  = wave * 128 + ct * 16 + l16;
            Out[((size_t)(b * SS + qrow)) * DD + col] = acc[ct][r] * linv[r];
        }
    }
}

// -------------------------------------------------------------------------
extern "C" void kernel_launch(void* const* d_in, const int* in_sizes, int n_in,
                              void* d_out, int out_size, void* d_ws, size_t ws_size,
                              hipStream_t stream) {
    (void)in_sizes; (void)n_in; (void)out_size; (void)ws_size;
    const float* q  = (const float*)d_in[0];
    const float* k  = (const float*)d_in[1];
    const float* v  = (const float*)d_in[2];
    // d_in[3] = mask (unused by reference forward)
    const float* Wq = (const float*)d_in[4];
    const float* bq = (const float*)d_in[5];
    const float* Wk = (const float*)d_in[6];
    const float* bk = (const float*)d_in[7];
    const float* Wv = (const float*)d_in[8];
    const float* bv = (const float*)d_in[9];

    const size_t NTOK = (size_t)BB * SS;          // 8192
    __bf16* Qp  = (__bf16*)d_ws;                  // 16 MB
    __bf16* Kp  = Qp + NTOK * DD;                 // 16 MB
    __bf16* VpT = Kp + NTOK * DD;                 // 16 MB
    __bf16* WqB = VpT + NTOK * DD;                // 2 MB
    __bf16* WkB = WqB + (size_t)DD * DD;          // 2 MB
    __bf16* WvB = WkB + (size_t)DD * DD;          // 2 MB  (total 54 MB of ws)

    // one-shot weight conversion (1M elements each -> 512 blocks)
    const int nW = DD * DD;
    cvt_f32_bf16<<<dim3(nW / 8 / 256), 256, 0, stream>>>(Wq, WqB, nW);
    cvt_f32_bf16<<<dim3(nW / 8 / 256), 256, 0, stream>>>(Wk, WkB, nW);
    cvt_f32_bf16<<<dim3(nW / 8 / 256), 256, 0, stream>>>(Wv, WvB, nW);

    // projections: grid (8192/128, 1024/64, 3) x 256 threads
    proj_kernel<<<dim3(64, 16, 3), 256, 0, stream>>>(q, k, v, WqB, bq, WkB, bk, WvB, bv,
                                                     Qp, Kp, VpT);
    // attention: grid (S/16 q-tiles, B) x 256 threads
    attn_kernel<<<dim3(SS / 16, BB), 256, 0, stream>>>(Qp, Kp, VpT, (float*)d_out);
}
